// PointTransformerLayer_2731599200752
// MI455X (gfx1250) — matrix operand order repaired and checked by reading, not compiled, not run on hardware
//
#include <hip/hip_runtime.h>
#include <hip/hip_bf16.h>

typedef __attribute__((ext_vector_type(16))) _Float16 v16h;
typedef __attribute__((ext_vector_type(8)))  _Float16 v8h;
typedef __attribute__((ext_vector_type(8)))  float    v8f;
typedef __attribute__((ext_vector_type(4)))  unsigned int u32x4;
typedef __attribute__((ext_vector_type(8)))  int i32x8;
typedef __attribute__((ext_vector_type(4)))  int i32x4;

#define C_DIM 128
#define TM    128      // rows per tile
#define LDT   136      // padded LDS row stride (halves): 16B aligned, bank-conflict free
#define BN_EPS 1e-5f
#define SM_EPS 1e-16f

#if __has_builtin(__builtin_amdgcn_tensor_load_to_lds) && __has_builtin(__builtin_amdgcn_s_wait_tensorcnt)
#define HAVE_TDM 1
#else
#define HAVE_TDM 0
#endif

// ---------------- tile load: global f16 [rows..128 x 128] -> LDS (stride LDT, zero-fill OOB rows) ----

__device__ __forceinline__ void load_tile_f16(_Float16* dst, const _Float16* src, int rows) {
#if HAVE_TDM
    if (threadIdx.x < 32) {   // wave 0 issues one TDM descriptor for the whole 128x128 tile
        unsigned lds_off = (unsigned)(uintptr_t)dst;          // generic LDS ptr: low 32 bits = LDS offset
        unsigned long long ga = (unsigned long long)(uintptr_t)src;
        // D# group0: count=1 | lds_addr | global_addr(57b) | type=2
        u32x4 g0 = { 1u, lds_off,
                     (unsigned)(ga & 0xffffffffu),
                     (unsigned)((ga >> 32) & 0x01ffffffu) | (2u << 30) };
        // D# group1: data_size=2B, pad_enable, pad_interval=5 (64 DW = one 128-half row),
        // pad_amount=3 (4 DW = 8 halves -> LDT=136); tensor 128 x rows; tile 128x128; stride0=128
        unsigned d1 = (unsigned)rows;
        i32x8 g1 = { (int)((1u << 16) | (1u << 20) | (5u << 22) | (3u << 25)),
                     (int)(128u << 16),                    // tensor_dim0[15:0]=128
                     (int)((d1 & 0xffffu) << 16),          // tensor_dim0_hi=0 | tensor_dim1[15:0]
                     (int)((d1 >> 16) | (128u << 16)),     // tensor_dim1_hi | tile_dim0=128
                     (int)128,                             // tile_dim1=128 | tile_dim2=0
                     (int)128,                             // tensor_dim0_stride[31:0]=128
                     (int)(16384u << 16),                  // stride0_hi=0 | tensor_dim1_stride[15:0]
                     0 };
        i32x4 gz = { 0, 0, 0, 0 };
#if __clang_major__ >= 23
        i32x8 gz8 = { 0, 0, 0, 0, 0, 0, 0, 0 };
        __builtin_amdgcn_tensor_load_to_lds(g0, g1, gz, gz, gz8, 0);
#else
        __builtin_amdgcn_tensor_load_to_lds(g0, g1, gz, gz, 0);
#endif
        __builtin_amdgcn_s_wait_tensorcnt(0);
    }
#else
    for (int i = threadIdx.x; i < 128 * 16; i += blockDim.x) {
        int r = i >> 4, c8 = (i & 15) << 3;
        v8h vv;
        if (r < rows) vv = *(const v8h*)(src + (long long)r * 128 + c8);
        else          vv = (v8h)((_Float16)0.0f);
        *(v8h*)(dst + r * LDT + c8) = vv;
    }
#endif
}

// ---------------- WMMA tile machinery (wave32, 16x16x32 f16 -> f32) ----------------

__device__ __forceinline__ v16h ldsA_frag(const _Float16* A, int mb, int kb, int lane) {
    // 16-bit A 16x32: lanes 0-15 row M=lane, K {0..7,16..23}; lanes 16-31 row M=lane-16, K {8..15,24..31}
    int l = lane & 15, hsel = (lane >> 4) & 1;
    const _Float16* p = A + (mb * 16 + l) * LDT + kb * 32 + hsel * 8;
    v8h lo = *(const v8h*)(p);
    v8h hi = *(const v8h*)(p + 16);
    return __builtin_shufflevector(lo, hi, 0,1,2,3,4,5,6,7,8,9,10,11,12,13,14,15);
}

__device__ __forceinline__ v16h ldsB_frag(const _Float16* B, int kb, int nb, int lane) {
    // 16-bit B 32x16: lane holds column N=lane%16, K {0..7,16..23} (lanes 0-15) / {8..15,24..31} (lanes 16-31)
    int l = lane & 15, hsel = (lane >> 4) & 1;
    const _Float16* p = B + (kb * 32 + hsel * 8) * LDT + nb * 16 + l;
    v16h r;
#pragma unroll
    for (int s = 0; s < 8; ++s) { r[s] = p[s * LDT]; r[s + 8] = p[(16 + s) * LDT]; }
    return r;
}

// Each wave (8 per block) computes rows [wave*16, wave*16+16) x all 128 cols of A(128x128)*B(128x128).
template <typename EPI>
__device__ __forceinline__ void wave_gemm_row(const _Float16* Alds, const _Float16* Blds, EPI&& epi) {
    const int lane = threadIdx.x & 31;
    const int wave = threadIdx.x >> 5;
    v16h a0 = ldsA_frag(Alds, wave, 0, lane);
    v16h a1 = ldsA_frag(Alds, wave, 1, lane);
    v16h a2 = ldsA_frag(Alds, wave, 2, lane);
    v16h a3 = ldsA_frag(Alds, wave, 3, lane);
#pragma unroll
    for (int nb = 0; nb < 8; ++nb) {
        v8f acc = {};
        acc = __builtin_amdgcn_wmma_f32_16x16x32_f16(false, a0, false, ldsB_frag(Blds, 0, nb, lane), (short)0, acc, false, false);
        acc = __builtin_amdgcn_wmma_f32_16x16x32_f16(false, a1, false, ldsB_frag(Blds, 1, nb, lane), (short)0, acc, false, false);
        acc = __builtin_amdgcn_wmma_f32_16x16x32_f16(false, a2, false, ldsB_frag(Blds, 2, nb, lane), (short)0, acc, false, false);
        acc = __builtin_amdgcn_wmma_f32_16x16x32_f16(false, a3, false, ldsB_frag(Blds, 3, nb, lane), (short)0, acc, false, false);
        epi(wave, nb, lane, acc);
    }
}

// ---------------- prep: f32->f16 weight conversion + BN folding ----------------

__global__ void ptl_prep_kernel(const float* W_in, const float* W_lin, const float* W_src, const float* W_dst,
                                const float* W_out, const float* W_a,
                                const float* g_a, const float* v_a, const float* b_a, const float* m_a, const float* be_a,
                                const float* W_p, const float* b_p, const float* g_p, const float* be_p,
                                const float* m_p, const float* v_p,
                                _Float16* win16, _Float16* wlin16, _Float16* wsrc16, _Float16* wdst16,
                                _Float16* wout16, _Float16* wa16,
                                float* wpf, float* bpf, float* bap) {
    const int MELEMS = 128 * 128;
    int id = blockIdx.x * blockDim.x + threadIdx.x;
    if (id < 6 * MELEMS) {
        int m = id / MELEMS, e = id % MELEMS;
        int n = e & 127;
        switch (m) {
            case 0: win16[e]  = (_Float16)W_in[e];  break;
            case 1: wlin16[e] = (_Float16)W_lin[e]; break;
            case 2: wsrc16[e] = (_Float16)W_src[e]; break;
            case 3: wdst16[e] = (_Float16)W_dst[e]; break;
            case 4: wout16[e] = (_Float16)W_out[e]; break;
            case 5: {
                float s = g_a[n] * rsqrtf(v_a[n] + BN_EPS);
                wa16[e] = (_Float16)(W_a[e] * s);
            } break;
        }
    } else {
        int c = id - 6 * MELEMS;
        if (c < 128) {
            float sp = g_p[c] * rsqrtf(v_p[c] + BN_EPS);
            wpf[c]       = W_p[c]       * sp;
            wpf[128 + c] = W_p[128 + c] * sp;
            wpf[256 + c] = W_p[256 + c] * sp;
            bpf[c] = (b_p[c] - m_p[c]) * sp + be_p[c];
            float sa = g_a[c] * rsqrtf(v_a[c] + BN_EPS);
            bap[c] = (b_a[c] - m_a[c]) * sa + be_a[c];
        }
    }
}

__global__ void ptl_zero_kernel(float* p, long long n) {
    long long i = (long long)blockIdx.x * blockDim.x + threadIdx.x;
    long long stride = (long long)gridDim.x * blockDim.x;
    for (; i < n; i += stride) p[i] = 0.0f;
}

// ---------------- node stage 1: h = relu(x @ W_in + b_in) -> f16 ----------------

__global__ void ptl_node_in_kernel(const float* x, const _Float16* win16, const float* b_in,
                                   _Float16* h16, int N) {
    __shared__ __align__(16) _Float16 As[TM * LDT];
    __shared__ __align__(16) _Float16 Bs[C_DIM * LDT];
    __shared__ float bias[C_DIM];
    load_tile_f16(Bs, win16, 128);
    if (threadIdx.x < 128) bias[threadIdx.x] = b_in[threadIdx.x];
    for (int tile = blockIdx.x; tile * TM < N; tile += gridDim.x) {
        const bool full = (tile * TM + TM) <= N;
        for (int i = threadIdx.x; i < TM * 32; i += blockDim.x) {
            int r = i >> 5, c4 = (i & 31) << 2;
            int n = tile * TM + r;
            float4 vv = (n < N) ? *(const float4*)(x + (long long)n * C_DIM + c4)
                                : make_float4(0.f, 0.f, 0.f, 0.f);
            _Float16* d = As + r * LDT + c4;
            d[0] = (_Float16)vv.x; d[1] = (_Float16)vv.y; d[2] = (_Float16)vv.z; d[3] = (_Float16)vv.w;
        }
        __syncthreads();
        wave_gemm_row(As, Bs, [&](int wave, int nb, int lane, v8f acc) {
            int l15 = lane & 15, hsel = (lane >> 4) & 1;
            int c = nb * 16 + l15;
            float bc = bias[c];
            int n0 = tile * TM + wave * 16 + hsel * 8;
            if (full) {
#pragma unroll
                for (int r = 0; r < 8; ++r) {
                    float v = acc[r] + bc;
                    v = v > 0.f ? v : 0.f;
                    h16[(long long)(n0 + r) * C_DIM + c] = (_Float16)v;
                }
            } else {
#pragma unroll
                for (int r = 0; r < 8; ++r) {
                    int n = n0 + r;
                    if (n < N) {
                        float v = acc[r] + bc;
                        v = v > 0.f ? v : 0.f;
                        h16[(long long)n * C_DIM + c] = (_Float16)v;
                    }
                }
            }
        });
        __syncthreads();
    }
}

// ---------------- node stage 2: v = h@W_lin (f32), a_src/a_dst = h@W_{src,dst} (f16) ----------------

__global__ void ptl_node_proj_kernel(const _Float16* h16,
                                     const _Float16* wlin16, const _Float16* wsrc16, const _Float16* wdst16,
                                     float* v32, _Float16* asrc16, _Float16* adst16, int N) {
    __shared__ __align__(16) _Float16 As[TM * LDT];
    __shared__ __align__(16) _Float16 Bs[C_DIM * LDT];
    for (int tile = blockIdx.x; tile * TM < N; tile += gridDim.x) {
        const bool full = (tile * TM + TM) <= N;
        load_tile_f16(As, h16 + (long long)tile * TM * C_DIM, N - tile * TM);
        __syncthreads();
        for (int w = 0; w < 3; ++w) {
            load_tile_f16(Bs, (w == 0) ? wlin16 : (w == 1) ? wsrc16 : wdst16, 128);
            __syncthreads();
            wave_gemm_row(As, Bs, [&](int wave, int nb, int lane, v8f acc) {
                int l15 = lane & 15, hsel = (lane >> 4) & 1;
                int c = nb * 16 + l15;
                int n0 = tile * TM + wave * 16 + hsel * 8;
#pragma unroll
                for (int r = 0; r < 8; ++r) {
                    int n = n0 + r;
                    if (full || n < N) {
                        long long o = (long long)n * C_DIM + c;
                        if (w == 0)      v32[o]    = acc[r];
                        else if (w == 1) asrc16[o] = (_Float16)acc[r];
                        else             adst16[o] = (_Float16)acc[r];
                    }
                }
            });
            __syncthreads();
        }
    }
}

// ---------------- edge passes: alpha GEMM + segment softmax (recompute, no E x C spill) ----------------

template <int PASS>
__global__ void ptl_edge_pass_kernel(const int* ei, const float* pos,
                                     const _Float16* asrc16, const _Float16* adst16, const float* v32,
                                     const _Float16* wa16, const float* bap_g, const float* wpf_g, const float* bpf_g,
                                     float* amax, float* denom, float* num, int E) {
    __shared__ __align__(16) _Float16 Ts[TM * LDT];
    __shared__ __align__(16) _Float16 Ws[C_DIM * LDT];
    __shared__ int sidx[TM], didx[TM];
    __shared__ float dp[TM][4];
    __shared__ float wp0[128], wp1[128], wp2[128], bp[128], ba[128];
    load_tile_f16(Ws, wa16, 128);
    if (threadIdx.x < 128) {
        int c = threadIdx.x;
        wp0[c] = wpf_g[c]; wp1[c] = wpf_g[128 + c]; wp2[c] = wpf_g[256 + c];
        bp[c] = bpf_g[c];  ba[c] = bap_g[c];
    }
    __syncthreads();
    const int ntiles = (E + TM - 1) / TM;
    for (int tile = blockIdx.x; tile < ntiles; tile += gridDim.x) {
        if (threadIdx.x == 0 && tile + (int)gridDim.x < ntiles) {
            __builtin_prefetch(ei + (long long)(tile + gridDim.x) * TM, 0, 1);
            __builtin_prefetch(ei + (long long)E + (long long)(tile + gridDim.x) * TM, 0, 1);
        }
        if (threadIdx.x < TM) {
            int e = tile * TM + threadIdx.x;
            int s = 0, d = 0;
            if (e < E) { s = ei[e]; d = ei[E + e]; }
            sidx[threadIdx.x] = s; didx[threadIdx.x] = d;
            dp[threadIdx.x][0] = pos[d * 3 + 0] - pos[s * 3 + 0];
            dp[threadIdx.x][1] = pos[d * 3 + 1] - pos[s * 3 + 1];
            dp[threadIdx.x][2] = pos[d * 3 + 2] - pos[s * 3 + 2];
        }
        __syncthreads();
        {   // build T[e][c] = a_dst[dst][c] - a_src[src][c] + delta(e,c)   (f16)
            int er = threadIdx.x >> 1;
            int half = threadIdx.x & 1;
            int s = sidx[er], d = didx[er];
            float p0 = dp[er][0], p1 = dp[er][1], p2 = dp[er][2];
            const _Float16* arow = adst16 + (long long)d * C_DIM;
            const _Float16* srow = asrc16 + (long long)s * C_DIM;
            for (int c0 = 0; c0 < 64; c0 += 8) {
                int c = half * 64 + c0;
                v8h av = *(const v8h*)(arow + c);
                v8h sv = *(const v8h*)(srow + c);
                _Float16* t = Ts + er * LDT + c;
#pragma unroll
                for (int j = 0; j < 8; ++j) {
                    int cc = c + j;
                    float del = fmaf(p0, wp0[cc], fmaf(p1, wp1[cc], fmaf(p2, wp2[cc], bp[cc])));
                    del = del > 0.f ? del : 0.f;
                    t[j] = (_Float16)((float)av[j] - (float)sv[j] + del);
                }
            }
        }
        __syncthreads();
        const bool full = (tile + 1) * TM <= E;   // uniform (always true for E % 128 == 0)
        wave_gemm_row(Ts, Ws, [&](int wave, int nb, int lane, v8f acc) {
            int l15 = lane & 15, hsel = (lane >> 4) & 1;
            int c = nb * 16 + l15;
            float bac = ba[c];
            float w0c = 0.f, w1c = 0.f, w2c = 0.f, bpc = 0.f;
            if (PASS == 2) { w0c = wp0[c]; w1c = wp1[c]; w2c = wp2[c]; bpc = bp[c]; }
            int er0 = wave * 16 + hsel * 8;
#pragma unroll
            for (int r = 0; r < 8; ++r) {
                int er = er0 + r;
                if (full || (tile * TM + er) < E) {
                    float alpha = acc[r] + bac;
                    alpha = alpha > 0.f ? alpha : 0.f;   // relu(BN-folded)
                    int d = didx[er];
                    if (PASS == 1) {
                        // non-negative floats: int-bit max == float max
                        atomicMax((int*)amax + (long long)d * C_DIM + c, __float_as_int(alpha));
                    } else {
                        float mx = amax[(long long)d * C_DIM + c];
                        float ev = __expf(alpha - mx);
                        int s = sidx[er];
                        float del = fmaf(dp[er][0], w0c, fmaf(dp[er][1], w1c, fmaf(dp[er][2], w2c, bpc)));
                        del = del > 0.f ? del : 0.f;
                        atomicAdd(denom + (long long)d * C_DIM + c, ev);
                        atomicAdd(num + (long long)d * C_DIM + c, ev * (v32[(long long)s * C_DIM + c] + del));
                    }
                }
            }
        });
        __syncthreads();
    }
}

// ---------------- node stage 3: out = relu((num/denom) @ W_out + b_out) ----------------

__global__ void ptl_node_out_kernel(const float* num, const float* denom,
                                    const _Float16* wout16, const float* b_out,
                                    float* out, int N) {
    __shared__ __align__(16) _Float16 As[TM * LDT];
    __shared__ __align__(16) _Float16 Bs[C_DIM * LDT];
    __shared__ float bias[C_DIM];
    load_tile_f16(Bs, wout16, 128);
    if (threadIdx.x < 128) bias[threadIdx.x] = b_out[threadIdx.x];
    for (int tile = blockIdx.x; tile * TM < N; tile += gridDim.x) {
        const bool full = (tile * TM + TM) <= N;
        for (int i = threadIdx.x; i < TM * 32; i += blockDim.x) {
            int r = i >> 5, c4 = (i & 31) << 2;
            int n = tile * TM + r;
            _Float16* d = As + r * LDT + c4;
            if (n < N) {
                long long o = (long long)n * C_DIM + c4;
                float4 nu = *(const float4*)(num + o);
                float4 de = *(const float4*)(denom + o);
                d[0] = (_Float16)(nu.x / (de.x + SM_EPS));
                d[1] = (_Float16)(nu.y / (de.y + SM_EPS));
                d[2] = (_Float16)(nu.z / (de.z + SM_EPS));
                d[3] = (_Float16)(nu.w / (de.w + SM_EPS));
            } else {
                d[0] = d[1] = d[2] = d[3] = (_Float16)0.0f;
            }
        }
        __syncthreads();
        wave_gemm_row(As, Bs, [&](int wave, int nb, int lane, v8f acc) {
            int l15 = lane & 15, hsel = (lane >> 4) & 1;
            int c = nb * 16 + l15;
            float bc = bias[c];
            int n0 = tile * TM + wave * 16 + hsel * 8;
            if (full) {
#pragma unroll
                for (int r = 0; r < 8; ++r) {
                    float v = acc[r] + bc;
                    v = v > 0.f ? v : 0.f;
                    out[(long long)(n0 + r) * C_DIM + c] = v;
                }
            } else {
#pragma unroll
                for (int r = 0; r < 8; ++r) {
                    int n = n0 + r;
                    if (n < N) {
                        float v = acc[r] + bc;
                        v = v > 0.f ? v : 0.f;
                        out[(long long)n * C_DIM + c] = v;
                    }
                }
            }
        });
        __syncthreads();
    }
}

// ---------------- launch ----------------

extern "C" void kernel_launch(void* const* d_in, const int* in_sizes, int n_in,
                              void* d_out, int out_size, void* d_ws, size_t ws_size,
                              hipStream_t stream) {
    const int N = in_sizes[0] / C_DIM;
    const int E = in_sizes[2] / 2;

    const float* x   = (const float*)d_in[0];
    const float* pos = (const float*)d_in[1];
    const int*   ei  = (const int*)d_in[2];
    const float* W_in = (const float*)d_in[3];
    const float* b_in = (const float*)d_in[4];
    const float* W_lin = (const float*)d_in[5];
    const float* W_src = (const float*)d_in[6];
    const float* W_dst = (const float*)d_in[7];
    const float* W_p  = (const float*)d_in[8];
    const float* b_p  = (const float*)d_in[9];
    const float* g_p  = (const float*)d_in[10];
    const float* be_p = (const float*)d_in[11];
    const float* m_p  = (const float*)d_in[12];
    const float* v_p  = (const float*)d_in[13];
    const float* W_a  = (const float*)d_in[14];
    const float* b_a  = (const float*)d_in[15];
    const float* g_a  = (const float*)d_in[16];
    const float* be_a = (const float*)d_in[17];
    const float* m_a  = (const float*)d_in[18];
    const float* v_a  = (const float*)d_in[19];
    const float* W_out = (const float*)d_in[20];
    const float* b_out = (const float*)d_in[21];
    float* out = (float*)d_out;

    // workspace carve-out
    char* p = (char*)d_ws;
    auto alloc = [&](size_t bytes) { void* r = (void*)p; p += (bytes + 255) & ~(size_t)255; return r; };
    const long long NC = (long long)N * C_DIM;
    _Float16* h16    = (_Float16*)alloc(NC * 2);
    _Float16* asrc16 = (_Float16*)alloc(NC * 2);
    _Float16* adst16 = (_Float16*)alloc(NC * 2);
    float*    v32    = (float*)alloc(NC * 4);
    float*    amax   = (float*)alloc(NC * 4 * 3);   // amax | denom | num contiguous
    float*    denom  = amax + NC;
    float*    num    = denom + NC;
    _Float16* win16  = (_Float16*)alloc(128 * 128 * 2);
    _Float16* wlin16 = (_Float16*)alloc(128 * 128 * 2);
    _Float16* wsrc16 = (_Float16*)alloc(128 * 128 * 2);
    _Float16* wdst16 = (_Float16*)alloc(128 * 128 * 2);
    _Float16* wout16 = (_Float16*)alloc(128 * 128 * 2);
    _Float16* wa16   = (_Float16*)alloc(128 * 128 * 2);
    float*    wpf    = (float*)alloc(3 * 128 * 4);
    float*    bpf    = (float*)alloc(128 * 4);
    float*    bap    = (float*)alloc(128 * 4);

    const int prep_elems = 6 * 128 * 128 + 128;
    ptl_prep_kernel<<<(prep_elems + 255) / 256, 256, 0, stream>>>(
        W_in, W_lin, W_src, W_dst, W_out, W_a,
        g_a, v_a, b_a, m_a, be_a,
        W_p, b_p, g_p, be_p, m_p, v_p,
        win16, wlin16, wsrc16, wdst16, wout16, wa16, wpf, bpf, bap);

    ptl_zero_kernel<<<2048, 256, 0, stream>>>(amax, NC * 3);

    const int node_tiles = (N + TM - 1) / TM;
    ptl_node_in_kernel<<<node_tiles, 256, 0, stream>>>(x, win16, b_in, h16, N);
    ptl_node_proj_kernel<<<node_tiles, 256, 0, stream>>>(h16, wlin16, wsrc16, wdst16, v32, asrc16, adst16, N);

    const int edge_tiles = (E + TM - 1) / TM;
    const int edge_grid = edge_tiles < 2048 ? edge_tiles : 2048;
    ptl_edge_pass_kernel<1><<<edge_grid, 256, 0, stream>>>(
        ei, pos, asrc16, adst16, v32, wa16, bap, wpf, bpf, amax, denom, num, E);
    ptl_edge_pass_kernel<2><<<edge_grid, 256, 0, stream>>>(
        ei, pos, asrc16, adst16, v32, wa16, bap, wpf, bpf, amax, denom, num, E);

    ptl_node_out_kernel<<<node_tiles, 256, 0, stream>>>(num, denom, wout16, b_out, out, N);
}
